// LinearE2V_wo_global_10290741641931
// MI455X (gfx1250) — matrix operand order repaired and checked by reading, not compiled, not run on hardware
//
#include <hip/hip_runtime.h>
#include <hip/hip_bf16.h>
#include <stdint.h>

// ---------------------------------------------------------------------------
// MI455X (gfx1250): bf16 WMMA (16x16x32, fp32 accum) for all matrix math.
// Incidence GEMM double-buffers its HBM stream into LDS with
// GLOBAL_LOAD_ASYNC_TO_LDS_B128 (ASYNCcnt); MLP weight tiles staged with the
// Tensor Data Mover (TENSORcnt) via the 6-arg clang-23 builtin.
// ---------------------------------------------------------------------------

typedef __bf16 bf16_t;
typedef __attribute__((ext_vector_type(16))) __bf16 v16bf;
typedef __attribute__((ext_vector_type(8)))  __bf16 v8bf;
typedef __attribute__((ext_vector_type(8)))  float  v8f;
typedef __attribute__((ext_vector_type(4)))  float  v4f;

#define DD 128
#define LN_EPS 1e-5f

// ---- async / tensor helpers ----------------------------------------------
static __device__ __forceinline__ void wait_async_le8() {
#if __has_builtin(__builtin_amdgcn_s_wait_asynccnt)
  __builtin_amdgcn_s_wait_asynccnt(8);
#else
  asm volatile("s_wait_asynccnt 0x8" ::: "memory");
#endif
}
static __device__ __forceinline__ void wait_async_le0() {
#if __has_builtin(__builtin_amdgcn_s_wait_asynccnt)
  __builtin_amdgcn_s_wait_asynccnt(0);
#else
  asm volatile("s_wait_asynccnt 0x0" ::: "memory");
#endif
}

// One 16-byte async copy global->LDS per lane (GVS addressing: sgpr base +
// per-lane 32-bit voffset; vdst VGPR carries the LDS byte address).
static __device__ __forceinline__ void async_cp16(uint32_t lds_off,
                                                  uint64_t gbase,
                                                  uint32_t goff) {
  asm volatile("global_load_async_to_lds_b128 %0, %1, %2"
               :
               : "v"(lds_off), "v"(goff), "s"(gbase)
               : "memory");
}

// ---- TDM weight staging (128x128 bf16 tile == tensor, 2D descriptor) ------
#if __has_builtin(__builtin_amdgcn_tensor_load_to_lds)
typedef unsigned int v4u   __attribute__((ext_vector_type(4)));
typedef int          v8i_t __attribute__((ext_vector_type(8)));
typedef int          v4i_t __attribute__((ext_vector_type(4)));

static __device__ __forceinline__ void tdm_load_w(bf16_t* dst, const bf16_t* src) {
  const uint64_t ga = (uint64_t)(uintptr_t)src;
  const uint32_t la = (uint32_t)(uintptr_t)dst;  // low 32 bits = LDS byte addr
  v4u g0;
  g0[0] = 1u;                                        // count=1 (valid user D#)
  g0[1] = la;                                        // lds_addr
  g0[2] = (uint32_t)ga;                              // global_addr[31:0]
  g0[3] = (uint32_t)((ga >> 32) & 0x1FFFFFFu) | (2u << 30);  // addr[56:32], type=2
  v8i_t g1;
  g1[0] = (int)(1u << 16);       // data_size=1 -> 2 bytes/elem; no multicast
  g1[1] = (int)(128u << 16);     // tensor_dim0 = 128
  g1[2] = (int)(128u << 16);     // tensor_dim1 = 128
  g1[3] = (int)(128u << 16);     // tile_dim0 = 128
  g1[4] = 128;                   // tile_dim1 = 128
  g1[5] = 128;                   // tensor_dim0_stride = 128
  g1[6] = 0;
  g1[7] = 0;
  v4i_t gz4 = {0, 0, 0, 0};
  v8i_t gz8 = {0, 0, 0, 0, 0, 0, 0, 0};
  // clang-23 / therock-10.0 arity: (g0, g1, g2, g3, g4, cpol)
  __builtin_amdgcn_tensor_load_to_lds(g0, g1, gz4, gz4, gz8, 0);
}
#endif

template <int TPB>
static __device__ __forceinline__ void load_w_lds(bf16_t* dst, const bf16_t* src) {
  const uint4* s = (const uint4*)src;
  uint4* d = (uint4*)dst;
#pragma unroll 4
  for (int i = threadIdx.x; i < (DD * DD * 2) / 16; i += TPB) d[i] = s[i];
}

template <int TPB>
static __device__ __forceinline__ void stage_weights(bf16_t* dst, const bf16_t* src) {
#if __has_builtin(__builtin_amdgcn_tensor_load_to_lds)
  if (threadIdx.x < 32) tdm_load_w(dst, src);   // one TDM descriptor, wave 0
#else
  load_w_lds<TPB>(dst, src);
#endif
}
static __device__ __forceinline__ void stage_weights_wait() {
#if __has_builtin(__builtin_amdgcn_tensor_load_to_lds)
  __builtin_amdgcn_s_wait_tensorcnt(0);
#endif
}

// ---- WMMA fragment loaders (wave32, 16x16x32 bf16) ------------------------
static __device__ __forceinline__ v16bf load_fragA(const bf16_t* base, int ld,
                                                   int r0, int k0) {
  const int lane = threadIdx.x & 31;
  const bf16_t* p = base + (size_t)(r0 + (lane & 15)) * ld + k0 + ((lane >> 4) << 3);
  v8bf lo = *(const v8bf*)(p);
  v8bf hi = *(const v8bf*)(p + 16);
  return __builtin_shufflevector(lo, hi, 0,1,2,3,4,5,6,7,8,9,10,11,12,13,14,15);
}

static __device__ __forceinline__ v16bf load_fragB(const bf16_t* base, int ld,
                                                   int n0, int k0) {
  const int lane = threadIdx.x & 31;
  const bf16_t* p = base + (size_t)(n0 + (lane & 15)) * ld + k0 + ((lane >> 4) << 3);
  v8bf lo = *(const v8bf*)(p);
  v8bf hi = *(const v8bf*)(p + 16);
  return __builtin_shufflevector(lo, hi, 0,1,2,3,4,5,6,7,8,9,10,11,12,13,14,15);
}

static __device__ __forceinline__ v16bf load_fragA_f32(const float* base, int ld,
                                                       int r0, int k0) {
  const int lane = threadIdx.x & 31;
  const float* p = base + (size_t)(r0 + (lane & 15)) * ld + k0 + ((lane >> 4) << 3);
  v4f a0 = *(const v4f*)(p + 0);
  v4f a1 = *(const v4f*)(p + 4);
  v4f a2 = *(const v4f*)(p + 16);
  v4f a3 = *(const v4f*)(p + 20);
  v16bf f;
#pragma unroll
  for (int i = 0; i < 4; ++i) {
    f[i]      = (bf16_t)a0[i];
    f[4 + i]  = (bf16_t)a1[i];
    f[8 + i]  = (bf16_t)a2[i];
    f[12 + i] = (bf16_t)a3[i];
  }
  return f;
}

static __device__ __forceinline__ void gemm_16x128_k128(const bf16_t* A,
                                                        const bf16_t* W,
                                                        int row0, v8f acc[8]) {
  const v8f zero = {0.f,0.f,0.f,0.f,0.f,0.f,0.f,0.f};
#pragma unroll
  for (int nt = 0; nt < 8; ++nt) acc[nt] = zero;
#pragma unroll
  for (int kc = 0; kc < 4; ++kc) {
    v16bf a = load_fragA(A, DD, row0, kc * 32);
#pragma unroll
    for (int nt = 0; nt < 8; ++nt) {
      v16bf b = load_fragB(W, DD, nt * 16, kc * 32);
      acc[nt] = __builtin_amdgcn_wmma_f32_16x16x32_bf16(
          false, a, false, b, (short)0, acc[nt], false, false);
    }
  }
}

static __device__ __forceinline__ void store_hidden_relu(v8f acc[8],
                                                         const float* b1,
                                                         bf16_t* H, int row0) {
  const int lane = threadIdx.x & 31;
  const int colb = lane & 15;
  const int m0 = row0 + ((lane >> 4) << 3);
#pragma unroll
  for (int nt = 0; nt < 8; ++nt) {
    const float bb = b1[nt * 16 + colb];
#pragma unroll
    for (int r = 0; r < 8; ++r) {
      float h = acc[nt][r] + bb;
      h = h > 0.f ? h : 0.f;
      H[(m0 + r) * DD + nt * 16 + colb] = (bf16_t)h;
    }
  }
}

// ---------------------------------------------------------------------------
// Kernel 0: fp32 -> bf16 weight conversion + transpose (W^T[n][k]) for 3 MLPs.
// ---------------------------------------------------------------------------
__global__ void convert_weights_kernel(
    const float* __restrict__ W1a, const float* __restrict__ W2a,
    const float* __restrict__ W1b, const float* __restrict__ W2b,
    const float* __restrict__ W1c, const float* __restrict__ W2c,
    bf16_t* __restrict__ w1t, bf16_t* __restrict__ w2t) {
  const int idx = blockIdx.x * 256 + threadIdx.x;
  if (idx >= 3 * DD * DD) return;
  const int m = idx / (DD * DD);
  const int r = idx % (DD * DD);
  const int n = r / DD;
  const int k = r % DD;
  const float* W1 = (m == 0) ? W1a : (m == 1) ? W1b : W1c;
  const float* W2 = (m == 0) ? W2a : (m == 1) ? W2b : W2c;
  w1t[idx] = (bf16_t)W1[k * DD + n];
  w2t[idx] = (bf16_t)W2[k * DD + n];
}

// ---------------------------------------------------------------------------
// Kernel 1: stage 1 — mask + LN(norm1) + PE + MLP1 + residual + mask for x_v
// and x_e; edge output stored transposed bf16 xeT[b][d][e].
// ---------------------------------------------------------------------------
__global__ __launch_bounds__(128) void stage1_kernel(
    const float* __restrict__ x_v, const float* __restrict__ x_e,
    const int* __restrict__ edge_orders,
    const unsigned char* __restrict__ node_mask,
    const unsigned char* __restrict__ edge_mask,
    const float* __restrict__ pe1,
    const float* __restrict__ g1, const float* __restrict__ bn1,
    const float* __restrict__ mb1, const float* __restrict__ mb2,
    const bf16_t* __restrict__ w1t, const bf16_t* __restrict__ w2t,
    float* __restrict__ xv_out, bf16_t* __restrict__ xeT_out) {
  __shared__ bf16_t lds_x[64 * DD];   // 16 KB
  __shared__ bf16_t lds_h[64 * DD];   // 16 KB
  __shared__ bf16_t lds_w[DD * DD];   // 32 KB

  const int blk = blockIdx.x;
  const bool is_node = blk < 256;
  const int lid = is_node ? blk : blk - 256;
  const int b = lid >> 5;
  const int r0 = (lid & 31) * 64;
  const float* X = (is_node ? x_v : x_e) + (size_t)b * 2048 * DD;
  const unsigned char* mask = (is_node ? node_mask : edge_mask) + (size_t)b * 2048;
  const int* ord = edge_orders + (size_t)b * 2048;

  const int tid = threadIdx.x;

  stage_weights<128>(lds_w, w1t);   // TDM overlaps with the LN phase below

  {  // LayerNorm + PE, 2 threads per row (64 cols each)
    const int r = tid >> 1;
    const int c0 = (tid & 1) * 64;
    const int grow = r0 + r;
    const float* xrow = X + (size_t)grow * DD;
    float v[64];
    float s = 0.f, s2 = 0.f;
#pragma unroll
    for (int i = 0; i < 16; ++i) {
      v4f t = *(const v4f*)(xrow + c0 + 4 * i);
#pragma unroll
      for (int j = 0; j < 4; ++j) {
        const float f = t[j];
        v[4 * i + j] = f;
        s += f;
        s2 += f * f;
      }
    }
    s += __shfl_xor(s, 1, 32);
    s2 += __shfl_xor(s2, 1, 32);
    const float mu = s * (1.f / 128.f);
    const float var = s2 * (1.f / 128.f) - mu * mu;
    const float rs = rsqrtf(var + LN_EPS);
    const float* pe = pe1 + (is_node ? 1 : ord[grow]) * DD;
#pragma unroll
    for (int i = 0; i < 64; ++i) {
      const int c = c0 + i;
      const float val = (v[i] - mu) * rs * g1[c] + bn1[c] + pe[c];
      lds_x[r * DD + c] = (bf16_t)val;
    }
  }
  stage_weights_wait();
  __syncthreads();

  const int warp = tid >> 5;
  const int row0 = warp * 16;
  v8f acc[8];
  gemm_16x128_k128(lds_x, lds_w, row0, acc);
  store_hidden_relu(acc, mb1, lds_h, row0);
  __syncthreads();
  stage_weights<128>(lds_w, w2t);
  stage_weights_wait();
  __syncthreads();
  gemm_16x128_k128(lds_h, lds_w, row0, acc);

  const int lane = tid & 31;
  const int colb = lane & 15;
  const int m0 = row0 + ((lane >> 4) << 3);
#pragma unroll
  for (int nt = 0; nt < 8; ++nt) {
    const int col = nt * 16 + colb;
    const float bb = mb2[col];
#pragma unroll
    for (int r = 0; r < 8; ++r) {
      const int grow = r0 + m0 + r;
      const float o = X[(size_t)grow * DD + col] + acc[nt][r] + bb;
      const bool mk = mask[grow] != 0;
      if (is_node) {
        xv_out[((size_t)b * 2048 + grow) * DD + col] = mk ? o : 0.f;
      } else {
        xeT_out[((size_t)b * DD + col) * 2048 + grow] = mk ? (bf16_t)o : (bf16_t)0.f;
      }
    }
  }
}

// ---------------------------------------------------------------------------
// Kernel 2: agg GEMM (K=2048) with double-buffered async LDS staging.
// Per stage: 64x32 fp32 incidence tile (8 KB) + 128x32 bf16 xeT tile (8 KB).
// ---------------------------------------------------------------------------
static __device__ __forceinline__ void issue_stage_copy(
    uint64_t ginc, uint64_t gbt, int kc, uint32_t linc, uint32_t lbt) {
  const int tid = threadIdx.x;
#pragma unroll
  for (int i = 0; i < 4; ++i) {            // incidence: 64 rows x 128 B
    const int c = tid + i * 128;           // 512 chunks of 16 B
    const int row = c >> 3;
    const int o16 = c & 7;
    async_cp16(linc + (uint32_t)(row * 128 + o16 * 16), ginc,
               (uint32_t)(row * 8192 + kc * 4 + o16 * 16));
  }
#pragma unroll
  for (int i = 0; i < 4; ++i) {            // xeT: 128 rows x 64 B
    const int c = tid + i * 128;
    const int row = c >> 2;
    const int o16 = c & 3;
    async_cp16(lbt + (uint32_t)(row * 64 + o16 * 16), gbt,
               (uint32_t)(row * 4096 + kc * 2 + o16 * 16));
  }
}

__global__ __launch_bounds__(128) void agg_kernel(
    const float* __restrict__ incidence, const bf16_t* __restrict__ xeT,
    const float* __restrict__ xv_upd, const float* __restrict__ suffix,
    float* __restrict__ x1) {
  __shared__ float  inc_buf[2][64 * 32];   // 2 x 8 KB
  __shared__ bf16_t bt_buf[2][128 * 32];   // 2 x 8 KB

  const int blk = blockIdx.x;
  const int b = blk >> 5;
  const int n0 = (blk & 31) * 64;
  const float* inc = incidence + ((size_t)b * 2048 + n0) * 2048;
  const bf16_t* bT = xeT + (size_t)b * DD * 2048;
  const uint64_t ginc = (uint64_t)(uintptr_t)inc;
  const uint64_t gbt = (uint64_t)(uintptr_t)bT;
  const uint32_t linc0 = (uint32_t)(uintptr_t)&inc_buf[0][0];
  const uint32_t linc1 = (uint32_t)(uintptr_t)&inc_buf[1][0];
  const uint32_t lbt0 = (uint32_t)(uintptr_t)&bt_buf[0][0];
  const uint32_t lbt1 = (uint32_t)(uintptr_t)&bt_buf[1][0];

  const int warp = threadIdx.x >> 5;
  const int row0 = warp * 16;

  v8f acc[8];
  const v8f zero = {0.f,0.f,0.f,0.f,0.f,0.f,0.f,0.f};
#pragma unroll
  for (int nt = 0; nt < 8; ++nt) acc[nt] = zero;

  issue_stage_copy(ginc, gbt, 0, linc0, lbt0);   // prologue: stage 0

  for (int s = 0; s < 64; ++s) {
    if (s + 1 < 64) {
      // kick off the next tile while this one is consumed by WMMAs
      issue_stage_copy(ginc, gbt, (s + 1) * 32,
                       ((s + 1) & 1) ? linc1 : linc0,
                       ((s + 1) & 1) ? lbt1 : lbt0);
      __builtin_prefetch(inc + (size_t)(s + 2) * 32, 0, 1);
      wait_async_le8();    // stage s complete; stage s+1 (8 ops/wave) in flight
    } else {
      wait_async_le0();
    }
    __syncthreads();

    const float* ib = inc_buf[s & 1];
    const bf16_t* bb = bt_buf[s & 1];
    v16bf a = load_fragA_f32(ib, 32, row0, 0);
#pragma unroll
    for (int nt = 0; nt < 8; ++nt) {
      v16bf bfb = load_fragB(bb, 32, nt * 16, 0);
      acc[nt] = __builtin_amdgcn_wmma_f32_16x16x32_bf16(
          false, a, false, bfb, (short)0, acc[nt], false, false);
    }
    __syncthreads();
  }

  const int lane = threadIdx.x & 31;
  const int colb = lane & 15;
  const int m0 = row0 + ((lane >> 4) << 3);
#pragma unroll
  for (int r = 0; r < 8; ++r) {
    const int n = n0 + m0 + r;
    const float sc = 1.f / (1.f + suffix[(size_t)b * 2048 + n]);
#pragma unroll
    for (int nt = 0; nt < 8; ++nt) {
      const int d = nt * 16 + colb;
      const size_t idx = ((size_t)b * 2048 + n) * DD + d;
      x1[idx] = xv_upd[idx] + acc[nt][r] * sc;
    }
  }
}

// ---------------------------------------------------------------------------
// Kernel 3: stage 3 — LN(norm2)+pe2 -> MLP2 -> residual -> mask ->
// LN(norm3) -> MLP3 -> residual + bias_b -> mask.
// ---------------------------------------------------------------------------
__global__ __launch_bounds__(64) void stage3_kernel(
    const float* __restrict__ x1, const unsigned char* __restrict__ node_mask,
    const float* __restrict__ pe2, const float* __restrict__ bias_b,
    const float* __restrict__ g2, const float* __restrict__ bn2,
    const float* __restrict__ g3, const float* __restrict__ bn3,
    const float* __restrict__ m2b1, const float* __restrict__ m2b2,
    const float* __restrict__ m3b1, const float* __restrict__ m3b2,
    const bf16_t* __restrict__ w1t2, const bf16_t* __restrict__ w2t2,
    const bf16_t* __restrict__ w1t3, const bf16_t* __restrict__ w2t3,
    float* __restrict__ out) {
  __shared__ bf16_t lds_x[32 * DD];   // 8 KB
  __shared__ bf16_t lds_h[32 * DD];   // 8 KB
  __shared__ bf16_t lds_w[DD * DD];   // 32 KB
  __shared__ float  lds_f[32 * DD];   // 16 KB

  const int blk = blockIdx.x;
  const int b = blk >> 6;
  const int n0 = (blk & 63) * 32;
  const float* X = x1 + ((size_t)b * 2048 + n0) * DD;
  const unsigned char* mask = node_mask + (size_t)b * 2048 + n0;
  const int tid = threadIdx.x;
  const int warp = tid >> 5;
  const int row0 = warp * 16;
  const int lane = tid & 31;
  const int colb = lane & 15;
  const int m0 = row0 + ((lane >> 4) << 3);

  stage_weights<64>(lds_w, w1t2);

  {  // LN(norm2) + pe2[1]
    const int r = tid >> 1;
    const int c0 = (tid & 1) * 64;
    const float* xrow = X + (size_t)r * DD;
    float v[64];
    float s = 0.f, s2 = 0.f;
#pragma unroll
    for (int i = 0; i < 16; ++i) {
      v4f t = *(const v4f*)(xrow + c0 + 4 * i);
#pragma unroll
      for (int j = 0; j < 4; ++j) {
        const float f = t[j];
        v[4 * i + j] = f;
        s += f;
        s2 += f * f;
      }
    }
    s += __shfl_xor(s, 1, 32);
    s2 += __shfl_xor(s2, 1, 32);
    const float mu = s * (1.f / 128.f);
    const float var = s2 * (1.f / 128.f) - mu * mu;
    const float rs = rsqrtf(var + LN_EPS);
    const float* pe = pe2 + DD;
#pragma unroll
    for (int i = 0; i < 64; ++i) {
      const int c = c0 + i;
      const float val = (v[i] - mu) * rs * g2[c] + bn2[c] + pe[c];
      lds_x[r * DD + c] = (bf16_t)val;
    }
  }
  stage_weights_wait();
  __syncthreads();

  v8f acc[8];
  gemm_16x128_k128(lds_x, lds_w, row0, acc);
  store_hidden_relu(acc, m2b1, lds_h, row0);
  __syncthreads();
  stage_weights<64>(lds_w, w2t2);
  stage_weights_wait();
  __syncthreads();
  gemm_16x128_k128(lds_h, lds_w, row0, acc);

#pragma unroll
  for (int nt = 0; nt < 8; ++nt) {
    const int col = nt * 16 + colb;
    const float bb = m2b2[col];
#pragma unroll
    for (int r = 0; r < 8; ++r) {
      const int lrow = m0 + r;
      const float y = X[(size_t)lrow * DD + col] + acc[nt][r] + bb;
      lds_f[lrow * DD + col] = (mask[lrow] != 0) ? y : 0.f;
    }
  }
  __syncthreads();

  {  // LN(norm3) from lds_f
    const int r = tid >> 1;
    const int c0 = (tid & 1) * 64;
    const float* xrow = lds_f + r * DD;
    float v[64];
    float s = 0.f, s2 = 0.f;
#pragma unroll
    for (int i = 0; i < 64; ++i) {
      const float f = xrow[c0 + i];
      v[i] = f;
      s += f;
      s2 += f * f;
    }
    s += __shfl_xor(s, 1, 32);
    s2 += __shfl_xor(s2, 1, 32);
    const float mu = s * (1.f / 128.f);
    const float var = s2 * (1.f / 128.f) - mu * mu;
    const float rs = rsqrtf(var + LN_EPS);
#pragma unroll
    for (int i = 0; i < 64; ++i) {
      const int c = c0 + i;
      const float val = (v[i] - mu) * rs * g3[c] + bn3[c];
      lds_x[r * DD + c] = (bf16_t)val;
    }
  }
  __syncthreads();
  stage_weights<64>(lds_w, w1t3);
  stage_weights_wait();
  __syncthreads();
  gemm_16x128_k128(lds_x, lds_w, row0, acc);
  store_hidden_relu(acc, m3b1, lds_h, row0);
  __syncthreads();
  stage_weights<64>(lds_w, w2t3);
  stage_weights_wait();
  __syncthreads();
  gemm_16x128_k128(lds_h, lds_w, row0, acc);

#pragma unroll
  for (int nt = 0; nt < 8; ++nt) {
    const int col = nt * 16 + colb;
    const float bb = m3b2[col] + bias_b[col];
#pragma unroll
    for (int r = 0; r < 8; ++r) {
      const int lrow = m0 + r;
      float o = lds_f[lrow * DD + col] + acc[nt][r] + bb;
      o = (mask[lrow] != 0) ? o : 0.f;
      out[((size_t)b * 2048 + n0 + lrow) * DD + col] = o;
    }
  }
}

// ---------------------------------------------------------------------------
extern "C" void kernel_launch(void* const* d_in, const int* in_sizes, int n_in,
                              void* d_out, int out_size, void* d_ws,
                              size_t ws_size, hipStream_t stream) {
  (void)in_sizes; (void)n_in; (void)out_size; (void)ws_size;

  const float* x_v = (const float*)d_in[0];
  const float* x_e = (const float*)d_in[1];
  const float* incidence = (const float*)d_in[2];
  const float* suffix = (const float*)d_in[3];
  const int* edge_orders = (const int*)d_in[4];
  const unsigned char* node_mask = (const unsigned char*)d_in[5];
  const unsigned char* edge_mask = (const unsigned char*)d_in[6];
  const float* pe1 = (const float*)d_in[7];
  const float* pe2 = (const float*)d_in[8];
  const float* bias_b = (const float*)d_in[9];
  const float* m1W1 = (const float*)d_in[10];
  const float* m1b1 = (const float*)d_in[11];
  const float* m1W2 = (const float*)d_in[12];
  const float* m1b2 = (const float*)d_in[13];
  const float* m2W1 = (const float*)d_in[14];
  const float* m2b1 = (const float*)d_in[15];
  const float* m2W2 = (const float*)d_in[16];
  const float* m2b2 = (const float*)d_in[17];
  const float* m3W1 = (const float*)d_in[18];
  const float* m3b1 = (const float*)d_in[19];
  const float* m3W2 = (const float*)d_in[20];
  const float* m3b2 = (const float*)d_in[21];
  const float* n1g = (const float*)d_in[22];
  const float* n1b = (const float*)d_in[23];
  const float* n2g = (const float*)d_in[24];
  const float* n2b = (const float*)d_in[25];
  const float* n3g = (const float*)d_in[26];
  const float* n3b = (const float*)d_in[27];

  char* ws = (char*)d_ws;
  bf16_t* w1t = (bf16_t*)(ws);
  bf16_t* w2t = (bf16_t*)(ws + 98304);
  float* xv_upd = (float*)(ws + 196608);
  bf16_t* xeT = (bf16_t*)(ws + 196608 + 8388608);
  float* x1 = (float*)(ws + 196608 + 8388608 + 4194304);

  convert_weights_kernel<<<192, 256, 0, stream>>>(m1W1, m1W2, m2W1, m2W2,
                                                  m3W1, m3W2, w1t, w2t);

  stage1_kernel<<<512, 128, 0, stream>>>(x_v, x_e, edge_orders, node_mask,
                                         edge_mask, pe1, n1g, n1b, m1b1, m1b2,
                                         w1t, w2t, xv_upd, xeT);

  agg_kernel<<<256, 128, 0, stream>>>(incidence, xeT, xv_upd, suffix, x1);

  stage3_kernel<<<512, 64, 0, stream>>>(
      x1, node_mask, pe2, bias_b, n2g, n2b, n3g, n3b, m2b1, m2b2, m3b1, m3b2,
      w1t + 16384, w2t + 16384, w1t + 2 * 16384, w2t + 2 * 16384,
      (float*)d_out);
}
